// ADSF_61495341744391
// MI455X (gfx1250) — matrix operand order repaired and checked by reading, not compiled, not run on hardware
//
#include <hip/hip_runtime.h>

typedef _Float16 half_t;
typedef __attribute__((ext_vector_type(8)))  _Float16 v8h;
typedef __attribute__((ext_vector_type(16))) _Float16 v16h;
typedef __attribute__((ext_vector_type(8)))  float    v8f;

#define N_NODES 2048
#define IN_FEAT 512
#define OUT_FEAT 256
#define NHEAD   4
#define NHF     64
#define NEG_SLOPE 0.2f

// ---------- WMMA fragment loaders (CDNA5 ISA 7.12.2 layouts, wave32) ----------
// A (16x32 f16, row-major source, row stride ld halves):
//   lane l: g=l/16, m=l%16 ; halves = row[k0+8g .. +7] and row[k0+8g+16 .. +7]
__device__ __forceinline__ v16h load_frag_A(const half_t* __restrict__ base,
                                            int ld, int row0, int k0, int lane) {
    int g = lane >> 4, m = lane & 15;
    const half_t* p = base + (size_t)(row0 + m) * ld + k0 + 8 * g;
    v8h lo = *(const v8h*)(p);
    v8h hi = *(const v8h*)(p + 16);
    return __builtin_shufflevector(lo, hi, 0,1,2,3,4,5,6,7,8,9,10,11,12,13,14,15);
}

// B (32x16 f16, column-major source i.e. BT[n][k], row stride ld halves):
//   lane l: g=l/16, n=l%16 ; halves = col_n[k0+8g .. +7] and col_n[k0+8g+16 .. +7]
__device__ __forceinline__ v16h load_frag_B(const half_t* __restrict__ bt,
                                            size_t ld, int col0, int k0, int lane) {
    int g = lane >> 4, n = lane & 15;
    const half_t* p = bt + (size_t)(col0 + n) * ld + k0 + 8 * g;
    v8h lo = *(const v8h*)(p);
    v8h hi = *(const v8h*)(p + 16);
    return __builtin_shufflevector(lo, hi, 0,1,2,3,4,5,6,7,8,9,10,11,12,13,14,15);
}

// ---------- K0a: f32 -> f16 convert of h ----------
__global__ void k_cvt_h(const float* __restrict__ h, half_t* __restrict__ h16, int n) {
    int t = blockIdx.x * blockDim.x + threadIdx.x;
    if (t < n) h16[t] = (half_t)h[t];
}

// ---------- K0b: W (512x256 f32) -> WT16 (256x512 f16) ----------
__global__ void k_cvt_wT(const float* __restrict__ W, half_t* __restrict__ wt) {
    int t = blockIdx.x * blockDim.x + threadIdx.x;
    if (t < IN_FEAT * OUT_FEAT) {
        int k = t / OUT_FEAT, n = t % OUT_FEAT;
        wt[(size_t)n * IN_FEAT + k] = (half_t)W[t];
    }
}

// ---------- K1: g = h @ W  ->  gT16 (256 x 2048, f16, feature-major) ----------
// block = 128 threads (4 waves), each wave computes one 16x16 tile of g.
__global__ __launch_bounds__(128) void k_gemm_g(const half_t* __restrict__ h16,
                                                const half_t* __restrict__ wt,
                                                half_t* __restrict__ gT) {
    int lane = threadIdx.x & 31;
    int wave = threadIdx.x >> 5;                 // 0..3
    int i0 = blockIdx.x * 16;                    // node rows
    int n0 = blockIdx.y * 64 + wave * 16;        // output feature cols
    v8f c = {};
    for (int k0 = 0; k0 < IN_FEAT; k0 += 32) {
        v16h a = load_frag_A(h16, IN_FEAT, i0, k0, lane);
        v16h b = load_frag_B(wt, (size_t)IN_FEAT, n0, k0, lane);
        c = __builtin_amdgcn_wmma_f32_16x16x32_f16(false, a, false, b, (short)0, c, false, false);
    }
    // C: c[v] = D[v + 8g][n] ; write transposed: gT[col][row], rows contiguous in v
    int g = lane >> 4, n = lane & 15;
    v8h o;
#pragma unroll
    for (int v = 0; v < 8; ++v) o[v] = (half_t)c[v];
    *(v8h*)(gT + (size_t)(n0 + n) * N_NODES + i0 + 8 * g) = o;
}

// ---------- K2: el/er = per-head dot(g_row, attn_w) ----------
__global__ void k_el_er(const half_t* __restrict__ gT, const float* __restrict__ aw,
                        float* __restrict__ el, float* __restrict__ er) {
    int j = blockIdx.x * blockDim.x + threadIdx.x;
    if (j >= N_NODES) return;
    for (int h = 0; h < NHEAD; ++h) {
        float accl = 0.f, accr = 0.f;
        for (int f = 0; f < NHF; ++f) {
            float gv = (float)gT[(size_t)(h * NHF + f) * N_NODES + j];
            accl += gv * aw[f];
            accr += gv * aw[NHF + f];
        }
        el[j * NHEAD + h] = accl;
        er[j * NHEAD + h] = accr;
    }
}

// ---------- K3: fused triple-softmax attention + aggregation ----------
// One block = 16 output rows, all 4 heads, all 64 features. 256 threads = 8 waves.
// Wave w: head = w/2, feature-half = w%2 -> two 16x16 output tiles.
__global__ __launch_bounds__(256) void k_gat(const float* __restrict__ adj,
                                             const float* __restrict__ s,
                                             const float* __restrict__ el,
                                             const float* __restrict__ er,
                                             const half_t* __restrict__ gT,
                                             float* __restrict__ out) {
    __shared__ float s_red[16][16][5];
    __shared__ float s_invZ1[16][NHEAD];
    __shared__ float s_invZs[16];
    __shared__ float s_el[16][NHEAD];
    __shared__ float s_invZ2[16][NHEAD];
    __shared__ alignas(16) half_t s_w16[NHEAD][16][32];   // A-fragments (exp weights)

    int t = threadIdx.x;
    int i0 = blockIdx.x * 16;
    int lane = t & 31;
    int wave = t >> 5;            // 0..7
    int head_w = wave >> 1;       // head this wave multiplies
    int fh = wave & 1;            // which 32-feature half
    int cbase = head_w * NHF + fh * 32;

    int i = t >> 4;               // row (0..15) this thread serves for VALU work
    int tj = t & 15;              // j-sublane

    if (t < 64) s_el[t >> 2][t & 3] = el[(size_t)(i0 + (t >> 2)) * NHEAD + (t & 3)];
    __syncthreads();

    float eli[NHEAD];
#pragma unroll
    for (int h = 0; h < NHEAD; ++h) eli[h] = s_el[i][h];

    const float* adjrow = adj + (size_t)(i0 + i) * N_NODES;
    const float* srow   = s   + (size_t)(i0 + i) * N_NODES;

    // ---- pass 1: Z1[i,h] = sum_unmasked exp(lrelu(el+er)), Zs[i] = sum_unmasked exp(s)
    float z1p[NHEAD] = {0.f, 0.f, 0.f, 0.f};
    float zsp = 0.f;
    for (int j = tj; j < N_NODES; j += 16) {
        float av = adjrow[j];
        if (av != 0.f) {
            float sv = srow[j];
            float4 e4 = *(const float4*)(er + (size_t)j * NHEAD);
            float erj[4] = {e4.x, e4.y, e4.z, e4.w};
#pragma unroll
            for (int h = 0; h < NHEAD; ++h) {
                float x = eli[h] + erj[h];
                x = x > 0.f ? x : NEG_SLOPE * x;
                z1p[h] += __expf(x);
            }
            zsp += __expf(sv);
        }
    }
#pragma unroll
    for (int h = 0; h < NHEAD; ++h) s_red[i][tj][h] = z1p[h];
    s_red[i][tj][4] = zsp;
    __syncthreads();
    if (t < 16) {
        for (int q = 0; q < 5; ++q) {
            float acc = 0.f;
            for (int k = 0; k < 16; ++k) acc += s_red[t][k][q];
            if (q < 4) s_invZ1[t][q] = 1.0f / acc;
            else       s_invZs[t]    = 1.0f / acc;
        }
    }
    __syncthreads();

    float iZ1[NHEAD];
#pragma unroll
    for (int h = 0; h < NHEAD; ++h) iZ1[h] = s_invZ1[i][h];
    float iZs = s_invZs[i];

    // ---- pass 2: stream j in chunks of 32; stage exp(a+s') in LDS; WMMA-accumulate
    v8f c0 = {}, c1 = {};
    float z2p[NHEAD] = {0.f, 0.f, 0.f, 0.f};

    for (int j0 = 0; j0 < N_NODES; j0 += 32) {
#pragma unroll
        for (int b = 0; b < 2; ++b) {
            int k = tj + 16 * b;
            int j = j0 + k;
            float av = adjrow[j];
            float sv = srow[j];
            float4 e4 = *(const float4*)(er + (size_t)j * NHEAD);
            float erj[4] = {e4.x, e4.y, e4.z, e4.w};
            float sp = (av != 0.f) ? __expf(sv) * iZs : 0.f;
#pragma unroll
            for (int h = 0; h < NHEAD; ++h) {
                float ev;
                if (av != 0.f) {
                    float x = eli[h] + erj[h];
                    x = x > 0.f ? x : NEG_SLOPE * x;
                    float a = __expf(x) * iZ1[h];
                    ev = __expf(a + sp);
                } else {
                    ev = 1.0f;   // exp(0): masked entries are dense in final softmax
                }
                z2p[h] += ev;
                s_w16[h][i][k] = (half_t)ev;
            }
        }
        __syncthreads();
        // WMMA: A from LDS (ISA A-layout), B straight from global gT16 (L2-resident)
        {
            int g = lane >> 4, m = lane & 15;
            const half_t* pa = &s_w16[head_w][m][8 * g];
            v8h alo = *(const v8h*)pa;
            v8h ahi = *(const v8h*)(pa + 16);
            v16h a = __builtin_shufflevector(alo, ahi,
                        0,1,2,3,4,5,6,7,8,9,10,11,12,13,14,15);
            v16h b0 = load_frag_B(gT, (size_t)N_NODES, cbase,      j0, lane);
            v16h b1 = load_frag_B(gT, (size_t)N_NODES, cbase + 16, j0, lane);
            c0 = __builtin_amdgcn_wmma_f32_16x16x32_f16(false, a, false, b0, (short)0, c0, false, false);
            c1 = __builtin_amdgcn_wmma_f32_16x16x32_f16(false, a, false, b1, (short)0, c1, false, false);
        }
        __syncthreads();
    }

    // ---- reduce Z2 and normalize
#pragma unroll
    for (int h = 0; h < NHEAD; ++h) s_red[i][tj][h] = z2p[h];
    __syncthreads();
    if (t < 16) {
        for (int h = 0; h < NHEAD; ++h) {
            float acc = 0.f;
            for (int k = 0; k < 16; ++k) acc += s_red[t][k][h];
            s_invZ2[t][h] = 1.0f / acc;
        }
    }
    __syncthreads();

    {
        int g = lane >> 4, n = lane & 15;
#pragma unroll
        for (int v = 0; v < 8; ++v) {
            int row = v + 8 * g;
            float sc = s_invZ2[row][head_w];
            out[(size_t)(i0 + row) * OUT_FEAT + cbase + n]      = c0[v] * sc;
            out[(size_t)(i0 + row) * OUT_FEAT + cbase + 16 + n] = c1[v] * sc;
        }
    }
}

// ---------- launch ----------
extern "C" void kernel_launch(void* const* d_in, const int* in_sizes, int n_in,
                              void* d_out, int out_size, void* d_ws, size_t ws_size,
                              hipStream_t stream) {
    (void)in_sizes; (void)n_in; (void)out_size; (void)ws_size;
    const float* h   = (const float*)d_in[0];
    const float* adj = (const float*)d_in[1];
    const float* s   = (const float*)d_in[2];
    const float* W   = (const float*)d_in[3];
    const float* aw  = (const float*)d_in[4];
    float* out = (float*)d_out;

    char* ws = (char*)d_ws;
    size_t off = 0;
    auto alloc = [&](size_t bytes) -> void* {
        off = (off + 255) & ~(size_t)255;
        void* p = ws + off;
        off += bytes;
        return p;
    };
    half_t* h16 = (half_t*)alloc((size_t)N_NODES * IN_FEAT * sizeof(half_t));
    half_t* wt  = (half_t*)alloc((size_t)IN_FEAT * OUT_FEAT * sizeof(half_t));
    half_t* gT  = (half_t*)alloc((size_t)OUT_FEAT * N_NODES * sizeof(half_t));
    float*  el  = (float*)alloc((size_t)N_NODES * NHEAD * sizeof(float));
    float*  er  = (float*)alloc((size_t)N_NODES * NHEAD * sizeof(float));

    k_cvt_h<<<(N_NODES * IN_FEAT + 255) / 256, 256, 0, stream>>>(h, h16, N_NODES * IN_FEAT);
    k_cvt_wT<<<(IN_FEAT * OUT_FEAT + 255) / 256, 256, 0, stream>>>(W, wt);
    k_gemm_g<<<dim3(N_NODES / 16, OUT_FEAT / 64), 128, 0, stream>>>(h16, wt, gT);
    k_el_er<<<(N_NODES + 255) / 256, 256, 0, stream>>>(gT, aw, el, er);
    k_gat<<<N_NODES / 16, 256, 0, stream>>>(adj, s, el, er, gT, out);
}
